// TransformerBlock_4432406249904
// MI455X (gfx1250) — matrix operand order repaired and checked by reading, not compiled, not run on hardware
//
#include <hip/hip_runtime.h>
#include <hip/hip_bf16.h>
#include <stdint.h>

// ---------------------------------------------------------------------------
// Types / WMMA helpers (CDNA5 gfx1250, wave32)
// ---------------------------------------------------------------------------
typedef _Float16 half_t;
typedef __attribute__((ext_vector_type(16))) _Float16 v16h;
typedef __attribute__((ext_vector_type(8)))  float    v8f;

union FragA { v16h v; uint32_t u[8]; };
union FragC { v8f  v; float    f[8]; };

__device__ __forceinline__ v8f wmma32(v16h a, v16h b, v8f c) {
  // D(16x16,f32) = A(16x32,f16) x B(32x16,f16) + C
  return __builtin_amdgcn_wmma_f32_16x16x32_f16(false, a, false, b, (short)0, c,
                                                false, false);
}

// Async global->LDS copy engine (CDNA5): 16 bytes per lane, ASYNCcnt-tracked.
__device__ __forceinline__ void async_load_b128(uint32_t lds_addr,
                                                const void* gaddr) {
  asm volatile("global_load_async_to_lds_b128 %0, %1, off"
               :
               : "v"(lds_addr), "v"((uint64_t)(uintptr_t)gaddr)
               : "memory");
}
__device__ __forceinline__ void wait_async0() {
  asm volatile("s_wait_asynccnt 0" ::: "memory");
}
__device__ __forceinline__ uint32_t lds_off(const void* p) {
  return (uint32_t)(uintptr_t)p;  // flat->LDS mapping uses addr[31:0]
}

// A-fragment (16x32 f16): lane l holds row l%16; K = {2j (+8 if lane>=16)} for
// j<4 and {2j+8 (+8 if lane>=16)} for j>=4, packed pairs (k,k+1) per dword.
__device__ __forceinline__ v16h load_a_frag(const half_t* base, int r0, int ld) {
  const int lane = threadIdx.x & 31;
  const int lr   = lane & 15;
  const int hf8  = (lane >> 4) << 3;
  const half_t* p = base + (size_t)(r0 + lr) * ld;
  FragA f;
#pragma unroll
  for (int j = 0; j < 8; ++j) {
    const int k = ((j < 4) ? (2 * j) : (2 * j + 8)) + hf8;
    f.u[j] = *reinterpret_cast<const uint32_t*>(p + k);
  }
  return f.v;
}

// B-fragment (32x16 f16) read from a K-transposed staging buffer BT[n][k]:
// lane l holds col l%16; K = {2j (+8 if lane>=16)} for j<4 and
// {2j+8 (+8 if lane>=16)} for j>=4, pairs (k,k+1) per dword.
__device__ __forceinline__ v16h load_b_frag(const half_t* baseT, int c0, int k0,
                                            int ld) {
  const int lane = threadIdx.x & 31;
  const int lr   = lane & 15;
  const int hf16 = (lane >> 4) << 3;
  const half_t* p = baseT + (size_t)(c0 + lr) * ld + k0;
  FragA f;
#pragma unroll
  for (int j = 0; j < 8; ++j)
    f.u[j] = *reinterpret_cast<const uint32_t*>(p + ((j < 4) ? (2 * j) : (2 * j + 8)) + hf16);
  return f.v;
}

__device__ __forceinline__ float gelu_tanh(float x) {
  const float c = 0.7978845608028654f;  // sqrt(2/pi)
  return 0.5f * x * (1.0f + tanhf(c * (x + 0.044715f * x * x * x)));
}

// ---------------------------------------------------------------------------
// Problem constants
// ---------------------------------------------------------------------------
constexpr int Bc = 2, Pc = 2048, Mc = 1024, Nh = 16, Hd = 64, Fc = 4096;
constexpr int ROWS = Bc * Pc;        // 4096
constexpr int QKVC = 3 * Mc;         // 3072

// ---------------------------------------------------------------------------
// Pack / cast kernels (weights fp32 -> f16 in WMMA-friendly layouts)
// ---------------------------------------------------------------------------
__global__ __launch_bounds__(256) void pack_qkv_w_kernel(
    const float* __restrict__ kq, const float* __restrict__ kk,
    const float* __restrict__ kv, half_t* __restrict__ w) {
  const int i = blockIdx.x * 256 + threadIdx.x;    // 0 .. 3*1048576-1
  const int sel = i >> 20;
  const int r = i & 1048575;
  const int n = r >> 16;                 // head
  const int m = (r >> 6) & 1023;         // model dim
  const int h = r & 63;                  // head dim
  const float* src = (sel == 0) ? kq : (sel == 1) ? kk : kv;
  // w[m][sel*1024 + n*64 + h] = src[n][m][h]
  w[(size_t)m * QKVC + sel * Mc + n * Hd + h] =
      (half_t)src[((size_t)n * Mc + m) * Hd + h];
}

__global__ __launch_bounds__(256) void pack_qkv_b_kernel(
    const float* __restrict__ bq, const float* __restrict__ bk,
    const float* __restrict__ bv, float* __restrict__ bb) {
  const int i = blockIdx.x * 256 + threadIdx.x;    // 0..3071
  const int sel = i >> 10;
  const int r = i & 1023;
  bb[i] = ((sel == 0) ? bq : (sel == 1) ? bk : bv)[r];
}

__global__ __launch_bounds__(256) void cast_f16_kernel(const float* __restrict__ in,
                                                       half_t* __restrict__ out,
                                                       int n) {
  const int i = blockIdx.x * 256 + threadIdx.x;
  if (i < n) out[i] = (half_t)in[i];
}

// ---------------------------------------------------------------------------
// LayerNorm: one 256-thread block per row of 1024 floats -> f16 row
// ---------------------------------------------------------------------------
__global__ __launch_bounds__(256) void ln_kernel(const float* __restrict__ xin,
                                                 const float* __restrict__ sc,
                                                 const float* __restrict__ bi,
                                                 half_t* __restrict__ out) {
  const int row = blockIdx.x, tid = threadIdx.x;
  const float4 v = reinterpret_cast<const float4*>(xin + (size_t)row * Mc)[tid];
  float s = v.x + v.y + v.z + v.w;
  float q = v.x * v.x + v.y * v.y + v.z * v.z + v.w * v.w;
#pragma unroll
  for (int off = 16; off > 0; off >>= 1) {
    s += __shfl_xor(s, off, 32);
    q += __shfl_xor(q, off, 32);
  }
  __shared__ float ssum[8], sqsum[8];
  if ((tid & 31) == 0) { ssum[tid >> 5] = s; sqsum[tid >> 5] = q; }
  __syncthreads();
  float S = 0.f, Q = 0.f;
#pragma unroll
  for (int i = 0; i < 8; ++i) { S += ssum[i]; Q += sqsum[i]; }
  const float mu   = S * (1.0f / (float)Mc);
  const float var  = Q * (1.0f / (float)Mc) - mu * mu;
  const float rinv = rsqrtf(var + 1e-5f);
  const float4 scv = reinterpret_cast<const float4*>(sc)[tid];
  const float4 biv = reinterpret_cast<const float4*>(bi)[tid];
  union { half_t h[4]; uint2 u; } o;
  o.h[0] = (half_t)((v.x - mu) * rinv * scv.x + biv.x);
  o.h[1] = (half_t)((v.y - mu) * rinv * scv.y + biv.y);
  o.h[2] = (half_t)((v.z - mu) * rinv * scv.z + biv.z);
  o.h[3] = (half_t)((v.w - mu) * rinv * scv.w + biv.w);
  reinterpret_cast<uint2*>(out + (size_t)row * Mc)[tid] = o.u;
}

// ---------------------------------------------------------------------------
// WMMA GEMM: C[ROWS x Nc] = A[ROWS x K](f16) * B[K x Nc](f16) + epilogue
// Block 256 thr (8 waves), tile 128x128, BK=32. Wave = 64x32 = 4x2 WMMA tiles.
// A tile staged via async global->LDS copies (ASYNCcnt); B tile staged with a
// transpose so both fragment gathers are contiguous ds_load_b128s.
// EPI 0: +bias -> f16 ; EPI 1: +bias+residual -> f32 ; EPI 2: +bias, gelu -> f16
// ---------------------------------------------------------------------------
constexpr int BM = 128, BN = 128, BK = 32, LDT = BK + 8;

template <int EPI>
__global__ __launch_bounds__(256) void gemm_wmma_kernel(
    const half_t* __restrict__ A, const half_t* __restrict__ B,
    const float* __restrict__ bias, const float* __restrict__ res,
    void* __restrict__ Cout, int Nc, int K) {
  __shared__ half_t As[BM * LDT];   // row-major [m][k]
  __shared__ half_t Bt[BN * LDT];   // transposed [n][k]
  const int tid = threadIdx.x;
  const int lane = tid & 31, wid = tid >> 5;
  const int lr = lane & 15, hf = lane >> 4;
  const int m0 = blockIdx.y * BM, n0 = blockIdx.x * BN;
  const int wm = (wid & 1) * 64, wn = (wid >> 1) * 32;

  FragC acc[4][2];
#pragma unroll
  for (int mt = 0; mt < 4; ++mt)
#pragma unroll
    for (int nt = 0; nt < 2; ++nt)
#pragma unroll
      for (int e = 0; e < 8; ++e) acc[mt][nt].f[e] = 0.0f;

  for (int kb = 0; kb < K; kb += BK) {
    // Stage A tile: 128x32 f16 via async copy engine (16B per lane, 2 rounds).
#pragma unroll
    for (int i = 0; i < 2; ++i) {
      const int e = tid + i * 256;
      const int row = e >> 2;
      const int kc = (e & 3) << 3;
      async_load_b128(lds_off(&As[row * LDT + kc]),
                      A + (size_t)(m0 + row) * K + kb + kc);
    }
    // Stage B tile transposed: read 32x128 row-major, write Bt[n][k]
#pragma unroll
    for (int i = 0; i < 2; ++i) {
      const int e = tid + i * 256;
      const int kr = e >> 4;
      const int nc = (e & 15) << 3;
      const uint4 d =
          *reinterpret_cast<const uint4*>(B + (size_t)(kb + kr) * Nc + n0 + nc);
      const half_t* dh = reinterpret_cast<const half_t*>(&d);
#pragma unroll
      for (int j2 = 0; j2 < 8; ++j2) Bt[(nc + j2) * LDT + kr] = dh[j2];
    }
    wait_async0();        // this wave's async LDS fills complete
    __syncthreads();      // all waves' fills visible

    v16h af[4];
#pragma unroll
    for (int mt = 0; mt < 4; ++mt) af[mt] = load_a_frag(As, wm + mt * 16, LDT);
    v16h bf[2];
#pragma unroll
    for (int nt = 0; nt < 2; ++nt) bf[nt] = load_b_frag(Bt, wn + nt * 16, 0, LDT);
#pragma unroll
    for (int mt = 0; mt < 4; ++mt)
#pragma unroll
      for (int nt = 0; nt < 2; ++nt)
        acc[mt][nt].v = wmma32(af[mt], bf[nt], acc[mt][nt].v);
    __syncthreads();
  }

  // Epilogue. C frag: lane holds col lr, rows j + 8*hf.
#pragma unroll
  for (int mt = 0; mt < 4; ++mt) {
#pragma unroll
    for (int nt = 0; nt < 2; ++nt) {
      const int gcol = n0 + wn + nt * 16 + lr;
      const float bval = bias[gcol];
#pragma unroll
      for (int j = 0; j < 8; ++j) {
        const int grow = m0 + wm + mt * 16 + j + hf * 8;
        const size_t idx = (size_t)grow * Nc + gcol;
        const float val = acc[mt][nt].f[j] + bval;
        if constexpr (EPI == 0) {
          reinterpret_cast<half_t*>(Cout)[idx] = (half_t)val;
        } else if constexpr (EPI == 1) {
          reinterpret_cast<float*>(Cout)[idx] = val + res[idx];
        } else {
          reinterpret_cast<half_t*>(Cout)[idx] = (half_t)gelu_tanh(val);
        }
      }
    }
  }
}

// ---------------------------------------------------------------------------
// Flash attention. qkv buffer: [ROWS][3072] f16, cols = {q|k|v}*1024 + n*64 + h.
// Grid (P/64, Nh, Bc); block 128 (4 waves); wave owns 16 q-rows.
// K chunk staged via async global->LDS; V chunk transposed manually.
// Mask (faithful to reference): KEEP key > query, else score = -1e10.
// ---------------------------------------------------------------------------
__global__ __launch_bounds__(128) void attn_kernel(const half_t* __restrict__ qkv,
                                                   half_t* __restrict__ z) {
  const int qt = blockIdx.x, n = blockIdx.y, b = blockIdx.z;
  const int tid = threadIdx.x, lane = tid & 31, wid = tid >> 5;
  const int lr = lane & 15, hf = lane >> 4;
  const int q0 = qt * 64 + wid * 16;

  __shared__ half_t Ks[32 * 72];        // [key][h]   (B frag for S = Q*K^T)
  __shared__ half_t Vt[64 * 40];        // [h][key]   (B frag for Z = P*V)
  __shared__ half_t Pscr[4 * 16 * 40];  // per-wave P staging (C->A relayout)
  half_t* Pw = Pscr + wid * 16 * 40;

  // Load Q fragments (K-dim = 64 -> two 16x32 A-frags), straight from global.
  v16h aq0, aq1;
  {
    const half_t* p = qkv + (size_t)(b * Pc + q0 + lr) * QKVC + n * Hd;
    FragA f0, f1;
    const int hf8 = hf << 3;
#pragma unroll
    for (int j = 0; j < 8; ++j) {
      const int k = ((j < 4) ? (2 * j) : (2 * j + 8)) + hf8;
      f0.u[j] = *reinterpret_cast<const uint32_t*>(p + k);
      f1.u[j] = *reinterpret_cast<const uint32_t*>(p + 32 + k);
    }
    aq0 = f0.v; aq1 = f1.v;
  }

  float mrow[8], lrow[8];
#pragma unroll
  for (int j = 0; j < 8; ++j) { mrow[j] = -__builtin_inff(); lrow[j] = 0.0f; }
  FragC o[4];
#pragma unroll
  for (int ht = 0; ht < 4; ++ht)
#pragma unroll
    for (int e = 0; e < 8; ++e) o[ht].f[e] = 0.0f;

  for (int c = 0; c < Pc / 32; ++c) {
    const int key0 = c * 32;
    // Stage K chunk via async copies; V chunk transposed through VGPRs.
#pragma unroll
    for (int i = 0; i < 2; ++i) {
      const int e = tid + i * 128;          // 0..255
      const int key = e >> 3;               // 0..31
      const int h8 = (e & 7) << 3;          // 0,8,..,56
      const size_t grow = (size_t)(b * Pc + key0 + key) * QKVC + n * Hd;
      async_load_b128(lds_off(&Ks[key * 72 + h8]), qkv + grow + Mc + h8);
      const uint4 vd = *reinterpret_cast<const uint4*>(qkv + grow + 2 * Mc + h8);
      const half_t* vh = reinterpret_cast<const half_t*>(&vd);
#pragma unroll
      for (int jj = 0; jj < 8; ++jj) Vt[(h8 + jj) * 40 + key] = vh[jj];
    }
    wait_async0();
    __syncthreads();

    // S tiles (16q x 16key) via 2 WMMAs each (h split 0-31 / 32-63).
    v8f s0v = {}, s1v = {};
    s0v = wmma32(aq0, load_b_frag(Ks, 0, 0, 72), s0v);
    s0v = wmma32(aq1, load_b_frag(Ks, 0, 32, 72), s0v);
    s1v = wmma32(aq0, load_b_frag(Ks, 16, 0, 72), s1v);
    s1v = wmma32(aq1, load_b_frag(Ks, 16, 32, 72), s1v);
    FragC S0, S1; S0.v = s0v; S1.v = s1v;

    const int keyA = key0 + lr;        // tile 0 key column for this lane
    const int keyB = key0 + 16 + lr;   // tile 1
    float p0a[8], p1a[8], corr[8];
#pragma unroll
    for (int j = 0; j < 8; ++j) {
      const int qrow = q0 + j + hf * 8;
      const float v0 = (keyA > qrow) ? S0.f[j] * 0.125f : -1e10f;
      const float v1 = (keyB > qrow) ? S1.f[j] * 0.125f : -1e10f;
      float t = fmaxf(v0, v1);
#pragma unroll
      for (int off = 8; off > 0; off >>= 1) t = fmaxf(t, __shfl_xor(t, off, 32));
      const float mn = fmaxf(mrow[j], t);
      corr[j] = __expf(mrow[j] - mn);
      const float p0 = __expf(v0 - mn);
      const float p1 = __expf(v1 - mn);
      float rs = p0 + p1;
#pragma unroll
      for (int off = 8; off > 0; off >>= 1) rs += __shfl_xor(rs, off, 32);
      lrow[j] = lrow[j] * corr[j] + rs;
      mrow[j] = mn;
      p0a[j] = p0; p1a[j] = p1;
    }
    // Rescale running output accumulators.
#pragma unroll
    for (int ht = 0; ht < 4; ++ht)
#pragma unroll
      for (int j = 0; j < 8; ++j) o[ht].f[j] *= corr[j];

    // Stage P (C-layout -> row-major LDS) for A-fragment reload.
#pragma unroll
    for (int j = 0; j < 8; ++j) {
      const int r = j + hf * 8;
      Pw[r * 40 + lr]      = (half_t)p0a[j];
      Pw[r * 40 + 16 + lr] = (half_t)p1a[j];
    }
    const v16h ap = load_a_frag(Pw, 0, 40);   // wave-private: DS in-order
#pragma unroll
    for (int ht = 0; ht < 4; ++ht)
      o[ht].v = wmma32(ap, load_b_frag(Vt, ht * 16, 0, 40), o[ht].v);
    __syncthreads();
  }

  // Finalize: z[b,q,n,h] = o / l   (f16 row-major [ROWS][1024])
#pragma unroll
  for (int j = 0; j < 8; ++j) {
    const float inv = 1.0f / lrow[j];
    const size_t rbase = (size_t)(b * Pc + q0 + j + hf * 8) * Mc + n * Hd;
#pragma unroll
    for (int ht = 0; ht < 4; ++ht)
      z[rbase + ht * 16 + lr] = (half_t)(o[ht].f[j] * inv);
  }
}

// ---------------------------------------------------------------------------
// Host launcher
// ---------------------------------------------------------------------------
extern "C" void kernel_launch(void* const* d_in, const int* in_sizes, int n_in,
                              void* d_out, int out_size, void* d_ws,
                              size_t ws_size, hipStream_t stream) {
  (void)in_sizes; (void)n_in; (void)out_size; (void)ws_size;
  const float* x   = (const float*)d_in[0];
  const float* g1  = (const float*)d_in[1];
  const float* be1 = (const float*)d_in[2];
  const float* kq  = (const float*)d_in[3];
  const float* bq  = (const float*)d_in[4];
  const float* kk  = (const float*)d_in[5];
  const float* bk  = (const float*)d_in[6];
  const float* kv  = (const float*)d_in[7];
  const float* bv  = (const float*)d_in[8];
  const float* ko  = (const float*)d_in[9];
  const float* bo  = (const float*)d_in[10];
  const float* g2  = (const float*)d_in[11];
  const float* be2 = (const float*)d_in[12];
  const float* w1  = (const float*)d_in[13];
  const float* b1  = (const float*)d_in[14];
  const float* w2  = (const float*)d_in[15];
  const float* b2  = (const float*)d_in[16];

  char* ws = (char*)d_ws;
  size_t off = 0;
  auto alloc = [&](size_t bytes) {
    char* p = ws + off;
    off += (bytes + 255) & ~(size_t)255;
    return p;
  };
  half_t* wqkv = (half_t*)alloc((size_t)Mc * QKVC * 2);   //  6 MB
  float*  bqkv = (float*) alloc((size_t)QKVC * 4);        // 12 KB
  half_t* wo   = (half_t*)alloc((size_t)Mc * Mc * 2);     //  2 MB
  half_t* w1h  = (half_t*)alloc((size_t)Mc * Fc * 2);     //  8 MB
  half_t* w2h  = (half_t*)alloc((size_t)Fc * Mc * 2);     //  8 MB
  half_t* x1   = (half_t*)alloc((size_t)ROWS * Mc * 2);   //  8 MB
  half_t* qkvb = (half_t*)alloc((size_t)ROWS * QKVC * 2); // 24 MB
  half_t* zb   = (half_t*)alloc((size_t)ROWS * Mc * 2);   //  8 MB
  float*  yb   = (float*) alloc((size_t)ROWS * Mc * 4);   // 16 MB
  half_t* y2   = (half_t*)alloc((size_t)ROWS * Mc * 2);   //  8 MB
  half_t* hb   = (half_t*)alloc((size_t)ROWS * Fc * 2);   // 32 MB

  // 1) Weight packing / casts (f32 -> f16, WMMA layouts)
  pack_qkv_w_kernel<<<3 * 1048576 / 256, 256, 0, stream>>>(kq, kk, kv, wqkv);
  pack_qkv_b_kernel<<<QKVC / 256, 256, 0, stream>>>(bq, bk, bv, bqkv);
  cast_f16_kernel<<<(Mc * Mc) / 256, 256, 0, stream>>>(ko, wo, Mc * Mc);
  cast_f16_kernel<<<(Mc * Fc) / 256, 256, 0, stream>>>(w1, w1h, Mc * Fc);
  cast_f16_kernel<<<(Fc * Mc) / 256, 256, 0, stream>>>(w2, w2h, Fc * Mc);

  // 2) LN1 -> x1 (f16)
  ln_kernel<<<ROWS, 256, 0, stream>>>(x, g1, be1, x1);

  // 3) Fused QKV projection: [4096x1024] x [1024x3072] (+bias, f16 out)
  gemm_wmma_kernel<0><<<dim3(QKVC / BN, ROWS / BM), 256, 0, stream>>>(
      x1, wqkv, bqkv, nullptr, qkvb, QKVC, Mc);

  // 4) Flash attention -> z (f16 [4096x1024])
  attn_kernel<<<dim3(Pc / 64, Nh, Bc), 128, 0, stream>>>(qkvb, zb);

  // 5) Output projection + bias + residual(x) -> y (f32)
  gemm_wmma_kernel<1><<<dim3(Mc / BN, ROWS / BM), 256, 0, stream>>>(
      zb, wo, bo, x, yb, Mc, Mc);

  // 6) LN2 -> y2 (f16)
  ln_kernel<<<ROWS, 256, 0, stream>>>(yb, g2, be2, y2);

  // 7) MLP up + gelu -> h (f16 [4096x4096])
  gemm_wmma_kernel<2><<<dim3(Fc / BN, ROWS / BM), 256, 0, stream>>>(
      y2, w1h, b1, nullptr, hb, Fc, Mc);

  // 8) MLP down + bias + residual(y) -> d_out (f32)
  gemm_wmma_kernel<1><<<dim3(Mc / BN, ROWS / BM), 256, 0, stream>>>(
      hb, w2h, b2, yb, (float*)d_out, Mc, Fc);
}
